// SRNN_85152021610758
// MI455X (gfx1250) — compile-verified, matrix operand order checked by reading, not run on hardware
//
#include <hip/hip_runtime.h>
#include <hip/hip_bf16.h>

// ---- problem sizes ----
constexpr int kB = 64, kL = 512, kF = 38, kH = 256, kZ = 16;
constexpr int kBL = kB * kL;          // 32768
constexpr int kG3 = 3 * kH;           // 768

typedef _Float16 f16;
typedef __attribute__((ext_vector_type(16))) _Float16 v16h;
typedef __attribute__((ext_vector_type(8)))  _Float16 v8h;
typedef __attribute__((ext_vector_type(8)))  float    v8f;

__device__ __forceinline__ float sigmoidf_(float v) { return 1.0f / (1.0f + expf(-v)); }
__device__ __forceinline__ float softplusf_(float v) { return (v > 20.0f) ? v : log1pf(expf(v)); }

// Load one 16-half WMMA fragment for the A/B operand.
// p must already point at  base + row*ldk + k0 + 8*(lane>=16)   (16B aligned)
__device__ __forceinline__ v16h load_frag(const f16* p) {
  v8h lo = *(const v8h*)(p);
  v8h hi = *(const v8h*)(p + 16);
  return __builtin_shufflevector(lo, hi, 0,1,2,3,4,5,6,7,8,9,10,11,12,13,14,15);
}

// Opaque zero offset: redefined by empty asm each time it executes, so loads
// using it are loop-variant (no LICM -> no register residency -> no spill),
// while the pointer's global address-space provenance is preserved
// (-> global_load_b128, LOADcnt only; not flat).
__device__ __forceinline__ int opaque_zero() {
  int z = 0;
  asm volatile("" : "+v"(z));
  return z;
}

// ------------------------------------------------------------------
// Register-blocked f16 WMMA GEMM: out[m,n] = sum_k A[m,k]*W[n,k] + bias[n]
// Each wave computes an (MB x NB) block of 16x16 tiles.
// ------------------------------------------------------------------
template <int MB, int NB>
__global__ void __launch_bounds__(256)
gemm16(const f16* __restrict__ A, const f16* __restrict__ W,
       const float* __restrict__ bias, float* __restrict__ out,
       int MTg, int NTg, int K, int lda, int ldw, int ldo) {
  const int wave = threadIdx.x >> 5;
  const int lane = threadIdx.x & 31;
  const long gtile = (long)blockIdx.x * 8 + wave;
  if (gtile >= (long)MTg * NTg) return;     // wave-uniform: EXEC stays all-1 for WMMA
  const int mg = (int)(gtile / NTg), ng = (int)(gtile % NTg);
  const int m0 = mg * MB * 16, n0 = ng * NB * 16;
  const int hsel = (lane >> 4) << 3;        // +8 element offset for upper half-wave

  const f16* ap[MB];
  const f16* wp[NB];
#pragma unroll
  for (int mb = 0; mb < MB; ++mb)
    ap[mb] = A + (long)(m0 + mb * 16 + (lane & 15)) * lda + hsel;
#pragma unroll
  for (int nb = 0; nb < NB; ++nb)
    wp[nb] = W + (long)(n0 + nb * 16 + (lane & 15)) * ldw + hsel;

  v8f acc[MB][NB];
#pragma unroll
  for (int mb = 0; mb < MB; ++mb)
#pragma unroll
    for (int nb = 0; nb < NB; ++nb)
      acc[mb][nb] = (v8f){0.f,0.f,0.f,0.f,0.f,0.f,0.f,0.f};

  for (int k0 = 0; k0 < K; k0 += 32) {
    v16h a[MB];
#pragma unroll
    for (int mb = 0; mb < MB; ++mb) a[mb] = load_frag(ap[mb] + k0);
#pragma unroll
    for (int nb = 0; nb < NB; ++nb) {
      v16h b = load_frag(wp[nb] + k0);
#pragma unroll
      for (int mb = 0; mb < MB; ++mb)
        acc[mb][nb] = __builtin_amdgcn_wmma_f32_16x16x32_f16(
            false, a[mb], false, b, (short)0, acc[mb][nb], false, false);
    }
  }

#pragma unroll
  for (int nb = 0; nb < NB; ++nb) {
    const int n = n0 + nb * 16 + (lane & 15);
    const float bv = bias ? bias[n] : 0.0f;
#pragma unroll
    for (int mb = 0; mb < MB; ++mb) {
#pragma unroll
      for (int r = 0; r < 8; ++r) {
        const int m = m0 + mb * 16 + r + ((lane >> 4) << 3);
        out[(long)m * ldo + n] = acc[mb][nb][r] + bv;
      }
    }
  }
}

// ------------------------------------------------------------------
// GRU recurrence.  One block = 16 batch rows; 16 waves (512 threads).
// Wave w owns output columns [16w,16w+16) of all three gates.
// - r/z gate weights register-resident (128 VGPRs); n-gate weights
//   streamed per step from L2 as ordinary global_load_b128
//   (opaque offset -> loop-variant address -> no LICM, no spill)
// - h state: f16 in LDS (double buffer, compile-time constant bases
//   -> ds_load_b128), f32 in registers
// - ONE barrier per step; hout HBM store issued after the barrier
// xi: [B, L, 768] f32 (b_ih folded in), whh: [768,256] f16, bhh: [768]
// ------------------------------------------------------------------
__global__ void __launch_bounds__(512)
gru_rec(const float* __restrict__ xi, const f16* __restrict__ whh,
        const float* __restrict__ bhh, float* __restrict__ hout, int reverse) {
  __shared__ f16 hbuf0[16 * 272];
  __shared__ f16 hbuf1[16 * 272];
  const int lane = threadIdx.x & 31;
  const int wave = threadIdx.x >> 5;
  const int b0 = blockIdx.x * 16;
  for (int i = threadIdx.x; i < 16 * 272; i += 512) hbuf0[i] = (f16)0.0f;
  __syncthreads();

  const int hsel = (lane >> 4) << 3;
  const int n    = wave * 16 + (lane & 15);            // owned column
  const int hoff = (lane & 15) * 272 + hsel;           // A-frag base offset (row = lane&15)
  const float bhr = bhh[n], bhz = bhh[256 + n], bhn = bhh[512 + n];

  // r/z gate weight fragments -> registers (2 gates x 8 k-iters = 128 VGPRs)
  v16h bwr[8], bwz[8];
  const f16* wpn = whh + (long)(512 + n) * 256 + hsel;   // n-gate rows (streamed)
  {
    const f16* wp0 = whh + (long)(      n) * 256 + hsel;
    const f16* wp1 = whh + (long)(256 + n) * 256 + hsel;
#pragma unroll
    for (int kk = 0; kk < 8; ++kk) {
      bwr[kk] = load_frag(wp0 + kk * 32);
      bwz[kk] = load_frag(wp1 + kk * 32);
    }
  }

  float hreg[8];
#pragma unroll
  for (int r = 0; r < 8; ++r) hreg[r] = 0.0f;

  auto gstep = [&](int tt, const f16* hsrc, f16* hdst) {
    const int t = reverse ? (kL - 1 - tt) : tt;
    // loop-variant view of the n-gate weights: loads stay inside the step,
    // but provenance stays with wpn -> global_load_b128
    const f16* wpn_t = wpn + opaque_zero();

    // gh = h @ whh^T for the owned columns of all 3 gates.
    // A fragments come from LDS (ds_load_b128), n-gate B from L2.
    v8f ar = {0.f,0.f,0.f,0.f,0.f,0.f,0.f,0.f};
    v8f az = ar, an = ar;
#pragma unroll
    for (int kk = 0; kk < 8; ++kk) {
      v16h ak = load_frag(hsrc + hoff + kk * 32);
      v16h bn = load_frag(wpn_t + kk * 32);
      ar = __builtin_amdgcn_wmma_f32_16x16x32_f16(false, ak, false, bwr[kk], (short)0, ar, false, false);
      az = __builtin_amdgcn_wmma_f32_16x16x32_f16(false, ak, false, bwz[kk], (short)0, az, false, false);
      an = __builtin_amdgcn_wmma_f32_16x16x32_f16(false, ak, false, bn,      (short)0, an, false, false);
    }

    // gate math; write next-step f16 state to LDS, keep f32 state in regs
#pragma unroll
    for (int r = 0; r < 8; ++r) {
      const int m = r + ((lane >> 4) << 3);
      const long xb = ((long)(b0 + m) * kL + t) * kG3;
      const float rg = sigmoidf_(xi[xb +       n] + ar[r] + bhr);
      const float zg = sigmoidf_(xi[xb + 256 + n] + az[r] + bhz);
      const float ng = tanhf   (xi[xb + 512 + n] + rg * (an[r] + bhn));
      const float hnew = (1.0f - zg) * ng + zg * hreg[r];
      hreg[r] = hnew;
      hdst[m * 272 + n] = (f16)hnew;
    }
    __syncthreads();     // only the ds_store needs to be visible -> dscnt wait

    // HBM store after the barrier: overlaps next step's WMMAs
#pragma unroll
    for (int r = 0; r < 8; ++r) {
      const int m = r + ((lane >> 4) << 3);
      hout[((long)(b0 + m) * kL + t) * kH + n] = hreg[r];
    }
  };

  for (int tt = 0; tt < kL; tt += 2) {
    gstep(tt,     hbuf0, hbuf1);   // compile-time constant LDS bases
    gstep(tt + 1, hbuf1, hbuf0);
  }
}

// ------------------------------------------------------------------
// Latent z recurrence (single block, 1024 threads = 64 batch x 16 z)
// Double-buffered z state -> one barrier per step.
// ------------------------------------------------------------------
__global__ void __launch_bounds__(1024)
zrec(const float* __restrict__ phi_pre, const float* __restrict__ wl,
     const float* __restrict__ ws_, const float* __restrict__ eps,
     float* __restrict__ zout, float* __restrict__ zlocout, float* __restrict__ zscaleout) {
  __shared__ float zsh[2][kB * kZ];
  const int tid = threadIdx.x, b = tid >> 4, zi = tid & 15;
  zsh[0][tid] = 0.0f;
  float wlz[16], wsz[16];
#pragma unroll
  for (int k = 0; k < 16; ++k) { wlz[k] = wl[zi * 272 + k]; wsz[k] = ws_[zi * 272 + k]; }
  __syncthreads();
  int cur = 0;
  for (int t = 0; t < kL; ++t) {
    const long pr = ((long)b * kL + t) * 32;
    float accl = phi_pre[pr + zi];
    float accs = phi_pre[pr + 16 + zi];
#pragma unroll
    for (int k = 0; k < 16; ++k) {
      const float zk = zsh[cur][b * 16 + k];
      accl += zk * wlz[k]; accs += zk * wsz[k];
    }
    const float scale = softplusf_(accs);
    const float znew  = accl + scale * eps[((long)t * kB + b) * kZ + zi];
    zsh[cur ^ 1][tid] = znew;
    const long o = ((long)b * kL + t) * kZ + zi;
    zout[o] = znew; zlocout[o] = accl; zscaleout[o] = scale;
    __syncthreads();
    cur ^= 1;
  }
}

// ---------------- pack / cast helpers ----------------
__global__ void pad_cast_w(const float* __restrict__ src, f16* __restrict__ dst,
                           int N, int K, int Kp) {
  long idx = (long)blockIdx.x * blockDim.x + threadIdx.x;
  if (idx >= (long)N * Kp) return;
  int k = (int)(idx % Kp), n = (int)(idx / Kp);
  dst[idx] = (f16)((k < K) ? src[(long)n * K + k] : 0.0f);
}

// stacked [w0;w1] -> [2*Nhalf, Kp] f16 + bias [2*Nhalf] f32; src rows stride srcLd, col offset kOff
__global__ void pack_pair(const float* __restrict__ w0, const float* __restrict__ b0,
                          const float* __restrict__ w1, const float* __restrict__ b1,
                          f16* __restrict__ dstW, float* __restrict__ dstB,
                          int N, int K, int Nhalf, int Kp, int kOff, int srcLd) {
  long idx = (long)blockIdx.x * blockDim.x + threadIdx.x;
  if (idx >= (long)2 * Nhalf * Kp) return;
  int k = (int)(idx % Kp), r = (int)(idx / Kp);
  int rr = (r < Nhalf) ? r : r - Nhalf;
  const float* ww = (r < Nhalf) ? w0 : w1;
  float v = (rr < N && k < K) ? ww[(long)rr * srcLd + kOff + k] : 0.0f;
  dstW[idx] = (f16)v;
  if (k == 0) {
    const float* bb = (r < Nhalf) ? b0 : b1;
    dstB[r] = (rr < N) ? bb[rr] : 0.0f;
  }
}

__global__ void build_xlag(const float* __restrict__ x, f16* __restrict__ dst) {
  long idx = (long)blockIdx.x * blockDim.x + threadIdx.x;
  if (idx >= (long)kBL * 64) return;
  int k = (int)(idx & 63); long row = idx >> 6; int l = (int)(row % kL);
  float v = (k < kF && l > 0) ? x[(row - 1) * kF + k] : 0.0f;
  dst[idx] = (f16)v;
}
__global__ void build_xh(const float* __restrict__ x, const float* __restrict__ h1,
                         f16* __restrict__ dst) {
  long idx = (long)blockIdx.x * blockDim.x + threadIdx.x;
  if (idx >= (long)kBL * 320) return;
  int k = (int)(idx % 320); long row = idx / 320;
  float v = 0.0f;
  if (k < kF) v = x[row * kF + k];
  else if (k < kF + kH) v = h1[row * kH + (k - kF)];
  dst[idx] = (f16)v;
}
__global__ void build_zh(const float* __restrict__ z, const float* __restrict__ h1,
                         f16* __restrict__ dst) {
  long idx = (long)blockIdx.x * blockDim.x + threadIdx.x;
  if (idx >= (long)kBL * 288) return;
  int k = (int)(idx % 288); long row = idx / 288;
  float v = 0.0f;
  if (k < kZ) v = z[row * kZ + k];
  else if (k < kZ + kH) v = h1[row * kH + (k - kZ)];
  dst[idx] = (f16)v;
}
__global__ void build_hz(const float* __restrict__ h1, const float* __restrict__ z,
                         f16* __restrict__ dst) {
  long idx = (long)blockIdx.x * blockDim.x + threadIdx.x;
  if (idx >= (long)kBL * 288) return;
  int k = (int)(idx % 288); long row = idx / 288; int l = (int)(row % kL);
  float v = 0.0f;
  if (k < kH) v = h1[row * kH + k];
  else if (k < kH + kZ) v = (l > 0) ? z[(row - 1) * kZ + (k - kH)] : 0.0f;
  dst[idx] = (f16)v;
}
__global__ void cast_f32_f16(const float* __restrict__ src, f16* __restrict__ dst, long n) {
  long idx = (long)blockIdx.x * blockDim.x + threadIdx.x;
  if (idx < n) dst[idx] = (f16)src[idx];
}
__global__ void init_out(float* out) {
  out[(long)kBL * kF] = 0.0f; out[(long)kBL * kF + 1] = 0.0f;
}

// ---------------- final heads eval + reductions ----------------
__global__ void __launch_bounds__(256)
finalize_kernel(const float* __restrict__ x, const float* __restrict__ yout,
                const float* __restrict__ zp, const float* __restrict__ zloc,
                const float* __restrict__ zscale, float* __restrict__ out) {
  const long idx = (long)blockIdx.x * 256 + threadIdx.x;   // grid covers exactly kBL
  const float HALF_LOG2PI = 0.9189385332046727f;
  float recon = 0.0f, kld = 0.0f;
#pragma unroll 2
  for (int f = 0; f < kF; ++f) {
    const float loc = yout[idx * 96 + f];
    const float sc  = softplusf_(yout[idx * 96 + 48 + f]);
    const float d   = (x[idx * kF + f] - loc) / sc;
    recon += 0.5f * d * d + logf(sc) + HALF_LOG2PI;
    out[idx * kF + f] = loc;
  }
#pragma unroll
  for (int zi = 0; zi < kZ; ++zi) {
    const float zs  = zscale[idx * kZ + zi];
    const float zl  = zloc[idx * kZ + zi];
    const float zpl = zp[idx * 32 + zi];
    const float zps = softplusf_(zp[idx * 32 + 16 + zi]);
    const float dd  = zl - zpl;
    kld += logf(zps / zs) + (zs * zs + dd * dd) / (2.0f * zps * zps) - 0.5f;
  }
  __shared__ float sr[256], sk[256];
  sr[threadIdx.x] = recon; sk[threadIdx.x] = kld;
  __syncthreads();
  for (int s = 128; s > 0; s >>= 1) {
    if (threadIdx.x < s) { sr[threadIdx.x] += sr[threadIdx.x + s]; sk[threadIdx.x] += sk[threadIdx.x + s]; }
    __syncthreads();
  }
  if (threadIdx.x == 0) {
    atomicAdd(out + (long)kBL * kF,     sr[0]);
    atomicAdd(out + (long)kBL * kF + 1, sk[0]);
  }
}

// ------------------------------------------------------------------
extern "C" void kernel_launch(void* const* d_in, const int* in_sizes, int n_in,
                              void* d_out, int out_size, void* d_ws, size_t ws_size,
                              hipStream_t stream) {
  (void)in_sizes; (void)n_in; (void)out_size; (void)ws_size;
  const float* x      = (const float*)d_in[0];
  const float* eps    = (const float*)d_in[1];
  const float* g1_wih = (const float*)d_in[2];
  const float* g1_whh = (const float*)d_in[3];
  const float* g1_bih = (const float*)d_in[4];
  const float* g1_bhh = (const float*)d_in[5];
  const float* g2_wih = (const float*)d_in[6];
  const float* g2_whh = (const float*)d_in[7];
  const float* g2_bih = (const float*)d_in[8];
  const float* g2_bhh = (const float*)d_in[9];
  const float* pz_lw  = (const float*)d_in[10];
  const float* pz_lb  = (const float*)d_in[11];
  const float* pz_sw  = (const float*)d_in[12];
  const float* pz_sb  = (const float*)d_in[13];
  const float* tz_lw  = (const float*)d_in[14];
  const float* tz_lb  = (const float*)d_in[15];
  const float* tz_sw  = (const float*)d_in[16];
  const float* tz_sb  = (const float*)d_in[17];
  const float* tx_lw  = (const float*)d_in[18];
  const float* tx_lb  = (const float*)d_in[19];
  const float* tx_sw  = (const float*)d_in[20];
  const float* tx_sb  = (const float*)d_in[21];
  float* out = (float*)d_out;

  char* wsb = (char*)d_ws;
  size_t off = 0;
  auto alloc = [&](size_t bytes) -> void* {
    void* p = wsb + off;
    off = (off + bytes + 255) & ~(size_t)255;
    return p;
  };

  // big region: xi1 -> xi2 -> (yout | zpout)
  float* big     = (float*)alloc((size_t)kBL * kG3 * 4);           // 96 MB
  f16*   xlag16  = (f16*)  alloc((size_t)kBL * 64 * 2);
  f16*   w1ih16  = (f16*)  alloc((size_t)kG3 * 64 * 2);
  f16*   whh1_16 = (f16*)  alloc((size_t)kG3 * kH * 2);
  f16*   whh2_16 = (f16*)  alloc((size_t)kG3 * kH * 2);
  f16*   w2ih16  = (f16*)  alloc((size_t)kG3 * 320 * 2);
  float* h1      = (float*)alloc((size_t)kBL * kH * 4);
  f16*   xh16    = (f16*)  alloc((size_t)kBL * 320 * 2);
  float* h2      = (float*)alloc((size_t)kBL * kH * 4);
  f16*   h2f16   = (f16*)  alloc((size_t)kBL * kH * 2);
  f16*   phi_h16 = (f16*)  alloc((size_t)32 * kH * 2);
  float* phi_b   = (float*)alloc(32 * 4);
  float* phi_pre = (float*)alloc((size_t)kBL * 32 * 4);
  float* zbuf    = (float*)alloc((size_t)kBL * kZ * 4);
  float* zloc    = (float*)alloc((size_t)kBL * kZ * 4);
  float* zscale  = (float*)alloc((size_t)kBL * kZ * 4);
  f16*   zh16    = (f16*)  alloc((size_t)kBL * 288 * 2);
  f16*   hz16    = (f16*)  alloc((size_t)kBL * 288 * 2);
  f16*   thx16   = (f16*)  alloc((size_t)96 * 288 * 2);
  float* thx_b   = (float*)alloc(96 * 4);
  f16*   thz16   = (f16*)  alloc((size_t)32 * 288 * 2);
  float* thz_b   = (float*)alloc(32 * 4);

  float* yout  = big;                                       // [BL, 96]
  float* zpout = (float*)((char*)big + ((size_t)32 << 20)); // [BL, 32], disjoint inside big

  auto g1d = [](long n) { return dim3((unsigned)((n + 255) / 256)); };
  auto gblk = [](long tiles) { return dim3((unsigned)((tiles + 7) / 8)); };
  const dim3 blk(256);

  // ---- weight packing ----
  pad_cast_w<<<g1d((long)kG3 * 64), blk, 0, stream>>>(g1_wih, w1ih16, kG3, kF, 64);
  pad_cast_w<<<g1d((long)kG3 * kH), blk, 0, stream>>>(g1_whh, whh1_16, kG3, kH, kH);
  pad_cast_w<<<g1d((long)kG3 * kH), blk, 0, stream>>>(g2_whh, whh2_16, kG3, kH, kH);
  pad_cast_w<<<g1d((long)kG3 * 320), blk, 0, stream>>>(g2_wih, w2ih16, kG3, kF + kH, 320);
  pack_pair<<<g1d((long)32 * 256), blk, 0, stream>>>(pz_lw, pz_lb, pz_sw, pz_sb,
      phi_h16, phi_b, 16, 256, 16, 256, 16, 272);
  pack_pair<<<g1d((long)96 * 288), blk, 0, stream>>>(tx_lw, tx_lb, tx_sw, tx_sb,
      thx16, thx_b, kF, 272, 48, 288, 0, 272);
  pack_pair<<<g1d((long)32 * 288), blk, 0, stream>>>(tz_lw, tz_lb, tz_sw, tz_sb,
      thz16, thz_b, 16, 272, 16, 288, 0, 272);

  // ---- GRU1: xi1 = lag(x) @ w1ih^T + b1ih ; recurrence ----
  build_xlag<<<g1d((long)kBL * 64), blk, 0, stream>>>(x, xlag16);
  gemm16<2, 4><<<gblk((long)(kBL / 32) * (kG3 / 64)), blk, 0, stream>>>(
      xlag16, w1ih16, g1_bih, big, kBL / 32, kG3 / 64, 64, 64, 64, kG3);
  gru_rec<<<dim3(kB / 16), dim3(512), 0, stream>>>(big, whh1_16, g1_bhh, h1, 0);

  // ---- GRU2 (reversed): xi2 = [x,h1] @ w2ih^T + b2ih ; recurrence ----
  build_xh<<<g1d((long)kBL * 320), blk, 0, stream>>>(x, h1, xh16);
  gemm16<2, 4><<<gblk((long)(kBL / 32) * (kG3 / 64)), blk, 0, stream>>>(
      xh16, w2ih16, g2_bih, big, kBL / 32, kG3 / 64, 320, 320, 320, kG3);
  gru_rec<<<dim3(kB / 16), dim3(512), 0, stream>>>(big, whh2_16, g2_bhh, h2, 1);

  // ---- latent recurrence: precompute h2 head contribution, then z scan ----
  cast_f32_f16<<<g1d((long)kBL * kH), blk, 0, stream>>>(h2, h2f16, (long)kBL * kH);
  gemm16<2, 2><<<gblk((long)(kBL / 32) * 1), blk, 0, stream>>>(
      h2f16, phi_h16, phi_b, phi_pre, kBL / 32, 1, kH, kH, kH, 32);
  zrec<<<dim3(1), dim3(1024), 0, stream>>>(phi_pre, pz_lw, pz_sw, eps, zbuf, zloc, zscale);

  // ---- heads: th_x on [z,h1], th_z on [h1, lag(z)] ----
  build_zh<<<g1d((long)kBL * 288), blk, 0, stream>>>(zbuf, h1, zh16);
  build_hz<<<g1d((long)kBL * 288), blk, 0, stream>>>(h1, zbuf, hz16);
  gemm16<2, 2><<<gblk((long)(kBL / 32) * 3), blk, 0, stream>>>(
      zh16, thx16, thx_b, yout, kBL / 32, 3, 288, 288, 288, 96);
  gemm16<2, 2><<<gblk((long)(kBL / 32) * 1), blk, 0, stream>>>(
      hz16, thz16, thz_b, zpout, kBL / 32, 1, 288, 288, 288, 32);

  // ---- outputs + reductions ----
  init_out<<<dim3(1), dim3(1), 0, stream>>>(out);
  finalize_kernel<<<dim3(kBL / 256), blk, 0, stream>>>(x, yout, zpout, zloc, zscale, out);
}
